// ForceConvolve_24833500906224
// MI455X (gfx1250) — matrix operand order repaired and checked by reading, not compiled, not run on hardware
//
#include <hip/hip_runtime.h>

// ---------------------------------------------------------------------------
// ForceConvolve for MI455X (gfx1250): bf16 WMMA GEMMs, LDS-fused 2-layer MLPs,
// f32 global atomics for segment sums.
// Register-blocked: each wave32 owns TWO 16-row A sub-tiles (32 output rows),
// so every B-fragment load pair feeds two independent v_wmma ops.
// ---------------------------------------------------------------------------

typedef unsigned short u16;
typedef u16    v4u   __attribute__((ext_vector_type(4)));
typedef u16    v8u   __attribute__((ext_vector_type(8)));
typedef u16    v16u  __attribute__((ext_vector_type(16)));
typedef __bf16 v16bf __attribute__((ext_vector_type(16)));
typedef float  v8f   __attribute__((ext_vector_type(8)));

#define HS 136                 // LDS row stride (elements): 272B, 16B-aligned, bank-spread
#define LOG2F_ 0.6931471805599453f

__device__ __forceinline__ u16 f2b(float f) {      // f32 -> bf16 (RNE)
  union { float f; unsigned u; } c; c.f = f;
  unsigned u = c.u;
  return (u16)((u + 0x7fffu + ((u >> 16) & 1u)) >> 16);
}
__device__ __forceinline__ float b2f(u16 h) {      // bf16 -> f32
  union { unsigned u; float f; } c; c.u = ((unsigned)h) << 16;
  return c.f;
}
__device__ __forceinline__ float ssp(float x) {    // softplus(x) - log(2), stable
  float t = __expf(-fabsf(x));
  return fmaxf(x, 0.0f) + __logf(1.0f + t) - LOG2F_;
}
__device__ __forceinline__ v16bf mkfrag(v8u lo, v8u hi) {
  v16u v = __builtin_shufflevector(lo, hi, 0,1,2,3,4,5,6,7,8,9,10,11,12,13,14,15);
  return __builtin_bit_cast(v16bf, v);
}

// ---------------------------------------------------------------------------
// Phase-1 GEMM: H = ssp(X @ W1 + b1), H (128 x 128, bf16) left in LDS.
// X: M x K1 bf16 row-major.  W1t: 128 x K1 bf16 (transposed weight, N-major).
// Block = 128 threads (4 wave32), each wave computes 32 rows (2 A sub-tiles).
// ---------------------------------------------------------------------------
template<int K1>
__device__ __forceinline__ void gemm_ssp_to_lds(
    const u16* __restrict__ X, long M, const u16* __restrict__ W1t,
    const float* __restrict__ b1, u16* Hl, long m0) {
  const int tid  = threadIdx.x;
  const int wave = tid >> 5;
  const int lane = tid & 31;
  const int half = lane >> 4;
  const int lr   = lane & 15;
  long row0 = m0 + wave * 32 + lr;
  long row1 = row0 + 16;
  if (row0 >= M) row0 = M - 1;               // clamp: garbage rows guarded at store
  if (row1 >= M) row1 = M - 1;
  const u16* x0 = X + row0 * (long)K1;
  const u16* x1 = X + row1 * (long)K1;

  v8f zero = {0.f,0.f,0.f,0.f,0.f,0.f,0.f,0.f};
  v8f acc[2][8];
#pragma unroll
  for (int nt = 0; nt < 8; ++nt) { acc[0][nt] = zero; acc[1][nt] = zero; }

#pragma unroll
  for (int kb = 0; kb < K1; kb += 32) {
    // A fragments (16x32 bf16): lane = row, K chunks per ISA layout
    v16bf a0 = mkfrag(*(const v8u*)(x0 + kb + half * 8),
                      *(const v8u*)(x0 + kb + 16 + half * 8));
    v16bf a1 = mkfrag(*(const v8u*)(x1 + kb + half * 8),
                      *(const v8u*)(x1 + kb + 16 + half * 8));
#pragma unroll
    for (int nt = 0; nt < 8; ++nt) {
      const u16* wr = W1t + (long)(nt * 16 + lr) * K1 + kb + half * 8;
      v16bf bfrag = mkfrag(*(const v8u*)wr, *(const v8u*)(wr + 16));
      acc[0][nt] = __builtin_amdgcn_wmma_f32_16x16x32_bf16(
          false, a0, false, bfrag, (short)0, acc[0][nt], false, false);
      acc[1][nt] = __builtin_amdgcn_wmma_f32_16x16x32_bf16(
          false, a1, false, bfrag, (short)0, acc[1][nt], false, false);
    }
  }
  // bias + ssp, stash hidden in LDS (C/D layout: reg r -> M = r + 8*half)
#pragma unroll
  for (int s = 0; s < 2; ++s) {
#pragma unroll
    for (int nt = 0; nt < 8; ++nt) {
      float bb = b1[nt * 16 + lr];
#pragma unroll
      for (int r = 0; r < 8; ++r) {
        float v = ssp(acc[s][nt][r] + bb);
        Hl[(wave * 32 + s * 16 + r + 8 * half) * HS + nt * 16 + lr] = f2b(v);
      }
    }
  }
}

// ---------------------------------------------------------------------------
// Fused 2-layer MLP:  Y = [Res +] ssp(X@W1+b1)@W2 + b2   (bf16 in/out, f32 acc)
// ---------------------------------------------------------------------------
template<int K1, bool RES>
__global__ __launch_bounds__(128, 1) void k_mlp2(
    const u16* __restrict__ X, long M,
    const u16* __restrict__ W1t, const float* __restrict__ b1,
    const u16* __restrict__ W2t, const float* __restrict__ b2,
    const u16* __restrict__ Res, u16* __restrict__ Y) {
  __shared__ u16 Hl[128 * HS];
  const long m0  = (long)blockIdx.x * 128;
  const int tid  = threadIdx.x;
  const int wave = tid >> 5, lane = tid & 31, half = lane >> 4, lr = lane & 15;

  gemm_ssp_to_lds<K1>(X, M, W1t, b1, Hl, m0);
  __syncthreads();

  v8f zero = {0.f,0.f,0.f,0.f,0.f,0.f,0.f,0.f};
  v8f acc[2][8];
#pragma unroll
  for (int nt = 0; nt < 8; ++nt) { acc[0][nt] = zero; acc[1][nt] = zero; }

  const u16* h0 = Hl + (wave * 32 + lr) * HS;       // wave reads its own rows
  const u16* h1 = h0 + 16 * HS;
#pragma unroll
  for (int kb = 0; kb < 128; kb += 32) {
    v16bf a0 = mkfrag(*(const v8u*)(h0 + kb + half * 8),
                      *(const v8u*)(h0 + kb + 16 + half * 8));
    v16bf a1 = mkfrag(*(const v8u*)(h1 + kb + half * 8),
                      *(const v8u*)(h1 + kb + 16 + half * 8));
#pragma unroll
    for (int nt = 0; nt < 8; ++nt) {
      const u16* wr = W2t + (long)(nt * 16 + lr) * 128 + kb + half * 8;
      v16bf bfrag = mkfrag(*(const v8u*)wr, *(const v8u*)(wr + 16));
      acc[0][nt] = __builtin_amdgcn_wmma_f32_16x16x32_bf16(
          false, a0, false, bfrag, (short)0, acc[0][nt], false, false);
      acc[1][nt] = __builtin_amdgcn_wmma_f32_16x16x32_bf16(
          false, a1, false, bfrag, (short)0, acc[1][nt], false, false);
    }
  }
  __syncthreads();
  // restage output tile (bias applied) into LDS for coalesced writeback
#pragma unroll
  for (int s = 0; s < 2; ++s) {
#pragma unroll
    for (int nt = 0; nt < 8; ++nt) {
      float bb = b2[nt * 16 + lr];
#pragma unroll
      for (int r = 0; r < 8; ++r)
        Hl[(wave * 32 + s * 16 + r + 8 * half) * HS + nt * 16 + lr] =
            f2b(acc[s][nt][r] + bb);
    }
  }
  __syncthreads();
  // coalesced b128 stores (+ optional residual add)
#pragma unroll
  for (int it = 0; it < 8; ++it) {
    int idx  = it * 128 + tid;       // 1024 chunks of 16 bf16
    int lrow = idx >> 3;
    int cc   = (idx & 7) * 16;
    long grow = m0 + lrow;
    if (grow < M) {
      const u16* hp = Hl + lrow * HS + cc;
      v8u v0 = *(const v8u*)hp, v1 = *(const v8u*)(hp + 8);
      u16* yp = Y + grow * 128 + cc;
      if (RES) {
        const u16* rp = Res + grow * 128 + cc;
        v8u r0 = *(const v8u*)rp, r1 = *(const v8u*)(rp + 8);
#pragma unroll
        for (int k = 0; k < 8; ++k) {
          v0[k] = f2b(b2f(v0[k]) + b2f(r0[k]));
          v1[k] = f2b(b2f(v1[k]) + b2f(r1[k]));
        }
      }
      *(v8u*)yp = v0; *(v8u*)(yp + 8) = v1;
    }
  }
}

// ---------------------------------------------------------------------------
// Readout: h = ssp(e@ro_W1+b1) (WMMA), scalar = h . ro_w2 + b2,
// f_edge = scalar * adj; atomically +into out[i], -into out[j].
// ---------------------------------------------------------------------------
__global__ __launch_bounds__(128, 1) void k_readout(
    const u16* __restrict__ e, const u16* __restrict__ W1t,
    const float* __restrict__ b1, const float* __restrict__ w2,
    const float* __restrict__ b2, const float* __restrict__ adj,
    const int* __restrict__ nbr, float* __restrict__ out, long E) {
  __shared__ u16 Hl[128 * HS];
  const long m0 = (long)blockIdx.x * 128;
  gemm_ssp_to_lds<128>(e, E, W1t, b1, Hl, m0);
  __syncthreads();
  int tid = threadIdx.x;
  long ge = m0 + tid;
  if (ge < E) {
    float d = b2[0];
    const u16* hr = Hl + tid * HS;
#pragma unroll 8
    for (int c = 0; c < 128; ++c) d += b2f(hr[c]) * w2[c];
    int i = nbr[2 * ge], j = nbr[2 * ge + 1];
#pragma unroll
    for (int c = 0; c < 3; ++c) {
      float fe = d * adj[ge * 3 + c];
      __hip_atomic_fetch_add(&out[(long)i * 3 + c],  fe,
                             __ATOMIC_RELAXED, __HIP_MEMORY_SCOPE_AGENT);
      __hip_atomic_fetch_add(&out[(long)j * 3 + c], -fe,
                             __ATOMIC_RELAXED, __HIP_MEMORY_SCOPE_AGENT);
    }
  }
}

// ---------------------------------------------------------------------------
// Edge geometry + Gaussian smearing (padded to 64 cols, bf16)
// ---------------------------------------------------------------------------
__global__ void k_edge(const float* __restrict__ xyz, const int* __restrict__ nbr,
                       float* __restrict__ adj, u16* __restrict__ e0, long E) {
  long t = (long)blockIdx.x * blockDim.x + threadIdx.x;
  long eid = t >> 6; int g = (int)(t & 63);
  if (eid >= E) return;
  int i = nbr[2 * eid], j = nbr[2 * eid + 1];
  float dx = xyz[3*(long)i]   - xyz[3*(long)j];
  float dy = xyz[3*(long)i+1] - xyz[3*(long)j+1];
  float dz = xyz[3*(long)i+2] - xyz[3*(long)j+2];
  float dis = sqrtf(dx*dx + dy*dy + dz*dz);
  if (g < 3) adj[eid*3 + g] = (g == 0 ? dx : (g == 1 ? dy : dz)) / dis;
  const float width = 5.0f / 49.0f;
  const float coeff = -0.5f / (width * width);
  float val = 0.0f;
  if (g < 50) { float dd = dis - width * (float)g; val = __expf(coeff * dd * dd); }
  e0[eid*64 + g] = f2b(val);
}

__global__ void k_embed(const float* __restrict__ emb, const int* __restrict__ z,
                        u16* __restrict__ r, long N) {
  long t = (long)blockIdx.x * blockDim.x + threadIdx.x;
  long n = t >> 7; int f = (int)(t & 127);
  if (n >= N) return;
  r[n*128 + f] = f2b(emb[(long)z[n]*128 + f]);
}

__global__ void k_zero(float* __restrict__ p, long n) {
  long t = (long)blockIdx.x * blockDim.x + threadIdx.x;
  if (t < n) p[t] = 0.0f;
}

__global__ void k_cvt(const float* __restrict__ s, u16* __restrict__ d, long n) {
  long t = (long)blockIdx.x * blockDim.x + threadIdx.x;
  if (t < n) d[t] = f2b(s[t]);
}

// rij = af[i]*ef, rji = af[j]*ef; s = rij+rji (bf16); f32 atomics into dr.
__global__ __launch_bounds__(256) void k_gather(
    const u16* __restrict__ af, const u16* __restrict__ ef,
    const int* __restrict__ nbr, u16* __restrict__ s,
    float* __restrict__ dr, long E) {
  long t = (long)blockIdx.x * 256 + threadIdx.x;
  long eid = t >> 5; int lane = (int)(t & 31);
  if (eid >= E) return;
  int i = nbr[2*eid], j = nbr[2*eid+1];
  int f0 = lane * 4;
  v4u efv = *(const v4u*)(ef + eid*128 + f0);
  v4u ai  = *(const v4u*)(af + (long)i*128 + f0);
  v4u aj  = *(const v4u*)(af + (long)j*128 + f0);
  v4u sv;
#pragma unroll
  for (int k = 0; k < 4; ++k) {
    float ev  = b2f(efv[k]);
    float rij = b2f(ai[k]) * ev;
    float rji = b2f(aj[k]) * ev;
    sv[k] = f2b(rij + rji);
    __hip_atomic_fetch_add(&dr[(long)j*128 + f0 + k], rij,
                           __ATOMIC_RELAXED, __HIP_MEMORY_SCOPE_AGENT);
    __hip_atomic_fetch_add(&dr[(long)i*128 + f0 + k], rji,
                           __ATOMIC_RELAXED, __HIP_MEMORY_SCOPE_AGENT);
  }
  *(v4u*)(s + eid*128 + f0) = sv;
}

// W (K x N, f32) -> Wt (N x Kpad, bf16), zero-padded in K
__global__ void k_transpose(const float* __restrict__ W, u16* __restrict__ Wt,
                            int K, int N, int Kpad) {
  long t = (long)blockIdx.x * blockDim.x + threadIdx.x;
  if (t >= (long)N * Kpad) return;
  int n = (int)(t / Kpad), k = (int)(t % Kpad);
  Wt[t] = (k < K) ? f2b(W[(long)k * N + n]) : (u16)0;
}

// ---------------------------------------------------------------------------
extern "C" void kernel_launch(void* const* d_in, const int* in_sizes, int n_in,
                              void* d_out, int out_size, void* d_ws, size_t ws_size,
                              hipStream_t stream) {
  const long N = 20000, E = 320000;
  const float* xyz    = (const float*)d_in[0];
  const int*   z      = (const int*)  d_in[1];
  const int*   nbr    = (const int*)  d_in[2];
  const float* emb    = (const float*)d_in[3];
  const float* ef_W1  = (const float*)d_in[4];
  const float* ef_b1  = (const float*)d_in[5];
  const float* ef_W2  = (const float*)d_in[6];
  const float* ef_b2  = (const float*)d_in[7];
  const float* cefW1  = (const float*)d_in[8];
  const float* cefb1  = (const float*)d_in[9];
  const float* cefW2  = (const float*)d_in[10];
  const float* cefb2  = (const float*)d_in[11];
  const float* cafW1  = (const float*)d_in[12];
  const float* cafb1  = (const float*)d_in[13];
  const float* cafW2  = (const float*)d_in[14];
  const float* cafb2  = (const float*)d_in[15];
  const float* cauW1  = (const float*)d_in[16];
  const float* caub1  = (const float*)d_in[17];
  const float* cauW2  = (const float*)d_in[18];
  const float* caub2  = (const float*)d_in[19];
  const float* ceuW1  = (const float*)d_in[20];
  const float* ceub1  = (const float*)d_in[21];
  const float* ceuW2  = (const float*)d_in[22];
  const float* ceub2  = (const float*)d_in[23];
  const float* ro_W1  = (const float*)d_in[24];
  const float* ro_b1  = (const float*)d_in[25];
  const float* ro_W2  = (const float*)d_in[26];
  const float* ro_b2  = (const float*)d_in[27];

  char* ws = (char*)d_ws;
  size_t off = 0;
  auto carve = [&](size_t bytes) -> void* {
    off = (off + 255) & ~(size_t)255;
    void* p = ws + off;
    off += bytes;
    return p;
  };
  const size_t WMAT = 128 * 128 * sizeof(u16);      // 32KB bf16 per 128x128
  u16* w_ef1  = (u16*)carve(128 * 64 * sizeof(u16));
  u16* w_ef2  = (u16*)carve(WMAT);
  u16* w_cef1 = (u16*)carve(3 * WMAT);
  u16* w_cef2 = (u16*)carve(3 * WMAT);
  u16* w_caf1 = (u16*)carve(3 * WMAT);
  u16* w_caf2 = (u16*)carve(3 * WMAT);
  u16* w_cau1 = (u16*)carve(3 * WMAT);
  u16* w_cau2 = (u16*)carve(3 * WMAT);
  u16* w_ceu1 = (u16*)carve(3 * WMAT);
  u16* w_ceu2 = (u16*)carve(3 * WMAT);
  u16* w_ro1  = (u16*)carve(WMAT);
  float* adj  = (float*)carve((size_t)E * 3 * sizeof(float));
  u16* e0b    = (u16*)carve((size_t)E * 64  * sizeof(u16));
  u16* eb     = (u16*)carve((size_t)E * 128 * sizeof(u16));
  u16* efb    = (u16*)carve((size_t)E * 128 * sizeof(u16));
  u16* sb     = (u16*)carve((size_t)E * 128 * sizeof(u16));
  u16* rb     = (u16*)carve((size_t)N * 128 * sizeof(u16));
  u16* afb    = (u16*)carve((size_t)N * 128 * sizeof(u16));
  u16* drb    = (u16*)carve((size_t)N * 128 * sizeof(u16));
  float* dracc = (float*)carve((size_t)N * 128 * sizeof(float));

  // --- weight prep (transpose + f32->bf16) -------------------------------
  k_transpose<<<(128*64 + 255)/256, 256, 0, stream>>>(ef_W1, w_ef1, 50, 128, 64);
  k_transpose<<<64, 256, 0, stream>>>(ef_W2, w_ef2, 128, 128, 128);
  for (int l = 0; l < 3; ++l) {
    size_t wo = (size_t)l * 128 * 128;
    k_transpose<<<64, 256, 0, stream>>>(cefW1 + wo, w_cef1 + wo, 128, 128, 128);
    k_transpose<<<64, 256, 0, stream>>>(cefW2 + wo, w_cef2 + wo, 128, 128, 128);
    k_transpose<<<64, 256, 0, stream>>>(cafW1 + wo, w_caf1 + wo, 128, 128, 128);
    k_transpose<<<64, 256, 0, stream>>>(cafW2 + wo, w_caf2 + wo, 128, 128, 128);
    k_transpose<<<64, 256, 0, stream>>>(cauW1 + wo, w_cau1 + wo, 128, 128, 128);
    k_transpose<<<64, 256, 0, stream>>>(cauW2 + wo, w_cau2 + wo, 128, 128, 128);
    k_transpose<<<64, 256, 0, stream>>>(ceuW1 + wo, w_ceu1 + wo, 128, 128, 128);
    k_transpose<<<64, 256, 0, stream>>>(ceuW2 + wo, w_ceu2 + wo, 128, 128, 128);
  }
  k_transpose<<<64, 256, 0, stream>>>(ro_W1, w_ro1, 128, 128, 128);

  // --- geometry, radial basis, embeddings --------------------------------
  k_edge<<<(unsigned)((E*64 + 255)/256), 256, 0, stream>>>(xyz, nbr, adj, e0b, E);
  const int gE = (int)(E / 128);            // 2500, exact
  const int gN = (int)((N + 127) / 128);    // 157
  k_mlp2<64, false><<<gE, 128, 0, stream>>>(e0b, E, w_ef1, ef_b1, w_ef2, ef_b2,
                                            nullptr, eb);
  k_embed<<<(unsigned)((N*128 + 255)/256), 256, 0, stream>>>(emb, z, rb, N);

  // --- message-passing layers --------------------------------------------
  for (int l = 0; l < 3; ++l) {
    size_t wo = (size_t)l * 128 * 128;
    size_t bo = (size_t)l * 128;
    k_mlp2<128, false><<<gE, 128, 0, stream>>>(eb, E, w_cef1 + wo, cefb1 + bo,
                                               w_cef2 + wo, cefb2 + bo, nullptr, efb);
    k_mlp2<128, false><<<gN, 128, 0, stream>>>(rb, N, w_caf1 + wo, cafb1 + bo,
                                               w_caf2 + wo, cafb2 + bo, nullptr, afb);
    k_zero<<<(unsigned)((N*128 + 255)/256), 256, 0, stream>>>(dracc, N * 128);
    k_gather<<<(unsigned)((E*32 + 255)/256), 256, 0, stream>>>(afb, efb, nbr, sb,
                                                               dracc, E);
    k_cvt<<<(unsigned)((N*128 + 255)/256), 256, 0, stream>>>(dracc, drb, N * 128);
    k_mlp2<128, true><<<gN, 128, 0, stream>>>(drb, N, w_cau1 + wo, caub1 + bo,
                                              w_cau2 + wo, caub2 + bo, rb, rb);
    k_mlp2<128, true><<<gE, 128, 0, stream>>>(sb, E, w_ceu1 + wo, ceub1 + bo,
                                              w_ceu2 + wo, ceub2 + bo, eb, eb);
  }

  // --- readout -----------------------------------------------------------
  k_zero<<<(unsigned)((N*3 + 255)/256), 256, 0, stream>>>((float*)d_out, N * 3);
  k_readout<<<gE, 128, 0, stream>>>(eb, w_ro1, ro_b1, ro_W2, ro_b2, adj, nbr,
                                    (float*)d_out, E);
}
